// ChaosViT_Anomaly_Detection_84593675862166
// MI455X (gfx1250) — compile-verified, matrix operand order checked by reading, not run on hardware
//
#include <hip/hip_runtime.h>
#include <hip/hip_bf16.h>

// ---------------------------------------------------------------------------
// ChaosViT forward on gfx1250 (MI455X). Compute-bound (~480 GFLOP of GEMM vs
// ~100 MB traffic @ 23.3 TB/s) => v_wmma_f32_16x16x32_f16 for all matmuls.
// Round 2: weights pre-transposed in the f32->f16 pass so ALL GEMM staging is
// contiguous 16B chunks, moved with GLOBAL_LOAD_ASYNC_TO_LDS_B128 (ASYNCcnt)
// into double-buffered LDS tiles; 8 WMMAs per barrier per wave (32x64 tile).
// ---------------------------------------------------------------------------

typedef __attribute__((ext_vector_type(16))) _Float16 v16h;
typedef __attribute__((ext_vector_type(8)))  _Float16 v8h;
typedef __attribute__((ext_vector_type(8)))  float    v8f;

#define TSEQ 197
#define TPAD 224          // multiple of 32 (WMMA K)
#define DMODEL 768
#define NHEAD 12
#define DHEAD 64
#define NLAYER 12
#define BATCH 16
#define DMLP 3072
#define NPATCH 196
#define KPATCH 2304       // 9*16*16
#define MPATCH_PAD 3200   // 3136 -> multiple of 128
#define MTOK (BATCH * TPAD)   // 3584 = 28*128
#define CHAOS_ITERS 5
#define LN_EPS 1e-5f

// --- CDNA5 async copy: 16B per lane, global -> LDS, tracked by ASYNCcnt -----
__device__ __forceinline__ unsigned lds_off(const void* p) {
  return (unsigned)(unsigned long long)p;  // flat shared addr[31:0] == LDS addr
}
__device__ __forceinline__ void async_cp16(unsigned lds_addr, const void* gaddr) {
  asm volatile("global_load_async_to_lds_b128 %0, %1, off"
               :: "v"(lds_addr), "v"(gaddr) : "memory");
}
__device__ __forceinline__ void wait_async0() {
  asm volatile("s_wait_asynccnt 0" ::: "memory");
}

// --- Fragment loaders -------------------------------------------------------
// CDNA5 16-bit A/Bt fragment layout (ISA 7.12.2): lanes 0-15 hold K {0..7,16..23},
// lanes 16-31 hold K {8..15,24..31} => two contiguous 8-half (16B) runs per lane.
__device__ __forceinline__ v16h frag_ld_h(const _Float16* rowbase, int kb) {
  v8h lo = *(const v8h*)(rowbase + kb);
  v8h hi = *(const v8h*)(rowbase + kb + 16);
  v16h f;
#pragma unroll
  for (int j = 0; j < 8; ++j) { f[j] = lo[j]; f[j + 8] = hi[j]; }
  return f;
}

__device__ __forceinline__ v16h frag_ld_f32(const float* rowbase, int kb) {
  v16h f;
#pragma unroll
  for (int j = 0; j < 8; ++j) f[j] = (_Float16)rowbase[kb + j];
#pragma unroll
  for (int j = 0; j < 8; ++j) f[j + 8] = (_Float16)rowbase[kb + 16 + j];
  return f;
}

__device__ __forceinline__ v8f wmma_f16(v16h a, v16h b, v8f c) {
  return __builtin_amdgcn_wmma_f32_16x16x32_f16(false, a, false, b, (short)0, c,
                                                false, false);
}

// --- WMMA GEMM: C[M,N] = op(A[M,K] @ Bt[N,K]^T + bias) (+resid) -------------
// A row-major f16 (lda=K), Bt row-major f16 (ldb=K, i.e. B pre-transposed).
// Block tile 128x128, 8 waves of 32x64, K-step 32, double-buffered async LDS.
#define GEMM_TM 128
#define GEMM_TN 128
#define GEMM_TK 32
#define LDSA 40           // padded K-stride in halves (80B rows, 16B aligned)
#define FLAG_GELU 1

__global__ __launch_bounds__(256) void gemm_wmma(
    const _Float16* __restrict__ A, int lda,
    const _Float16* __restrict__ Bt, int ldb,
    const float* __restrict__ bias,
    const float* __restrict__ resid,
    float* __restrict__ Cf32,
    _Float16* __restrict__ Cf16,
    int M, int N, int K, int flags) {
  __shared__ __align__(16) _Float16 As[2][GEMM_TM * LDSA];
  __shared__ __align__(16) _Float16 Bs[2][GEMM_TN * LDSA];
  (void)M;
  const int tid = threadIdx.x;
  const int lane = tid & 31;
  const int wid = tid >> 5;
  const int wm = wid & 3;          // 4 x 32 rows
  const int wn = wid >> 2;         // 2 x 64 cols
  const int hlf = lane >> 4;
  const int l16 = lane & 15;
  const int kb = hlf * 8;
  const int m0 = blockIdx.y * GEMM_TM;
  const int n0 = blockIdx.x * GEMM_TN;

  // staging map: 128 rows, 2 threads/row, 16 halves (2x16B asyncs) each
  const int sr = tid >> 1;
  const int sc = (tid & 1) * 16;

  auto stage = [&](int buf, int k0) {
    const _Float16* sa = A  + (size_t)(m0 + sr) * lda + k0 + sc;
    const _Float16* sb = Bt + (size_t)(n0 + sr) * ldb + k0 + sc;
    async_cp16(lds_off(&As[buf][sr * LDSA + sc]), sa);
    async_cp16(lds_off(&As[buf][sr * LDSA + sc + 8]), sa + 8);
    async_cp16(lds_off(&Bs[buf][sr * LDSA + sc]), sb);
    async_cp16(lds_off(&Bs[buf][sr * LDSA + sc + 8]), sb + 8);
  };

  v8f acc[2][4] = {};
  stage(0, 0);
  int buf = 0;
  for (int k0 = 0; k0 < K; k0 += GEMM_TK) {
    wait_async0();       // current buffer's async transfers complete (this wave)
    __syncthreads();     // ..and everyone else's; prev compute done before reuse
    if (k0 + GEMM_TK < K) stage(buf ^ 1, k0 + GEMM_TK);

    v16h af[2], bf[4];
#pragma unroll
    for (int f = 0; f < 2; ++f)
      af[f] = frag_ld_h(&As[buf][(wm * 32 + f * 16 + l16) * LDSA], kb);
#pragma unroll
    for (int f = 0; f < 4; ++f)
      bf[f] = frag_ld_h(&Bs[buf][(wn * 64 + f * 16 + l16) * LDSA], kb);
#pragma unroll
    for (int fm = 0; fm < 2; ++fm)
#pragma unroll
      for (int fn = 0; fn < 4; ++fn)
        acc[fm][fn] = wmma_f16(af[fm], bf[fn], acc[fm][fn]);
    buf ^= 1;
  }

  // Epilogue. C/D layout: VGPR v -> M = v + 8*(lane>=16), N = lane&15.
#pragma unroll
  for (int fm = 0; fm < 2; ++fm)
#pragma unroll
    for (int fn = 0; fn < 4; ++fn)
#pragma unroll
      for (int v = 0; v < 8; ++v) {
        int m = m0 + wm * 32 + fm * 16 + hlf * 8 + v;
        int n = n0 + wn * 64 + fn * 16 + l16;
        float val = acc[fm][fn][v];
        if (bias) val += bias[n];
        if (flags & FLAG_GELU) val = 0.5f * val * (1.0f + erff(val * 0.70710678f));
        if (resid) val += resid[(size_t)m * N + n];
        if (Cf32) Cf32[(size_t)m * N + n] = val;
        if (Cf16) Cf16[(size_t)m * N + n] = (_Float16)val;
      }
}

// --- Fused attention: S=QK^T/8 -> softmax -> chaos -> softmax -> P@V --------
// One workgroup (256 thr / 8 waves) per (b,h). Dynamic LDS ~295 KB (< 320 KB/WGP).
#define QS_STRIDE 72
#define VT_STRIDE 232
#define S_STRIDE  TPAD
#define SM_QS ((size_t)TPAD * QS_STRIDE * 2)
#define SM_KS SM_QS
#define SM_VT ((size_t)DHEAD * VT_STRIDE * 2)
#define SM_S  ((size_t)TPAD * TPAD * 4)
#define SMEM_ATTN (SM_QS + SM_KS + SM_VT + SM_S)

__global__ __launch_bounds__(256) void attn_chaos(
    const _Float16* __restrict__ Q, const _Float16* __restrict__ Kmat,
    const _Float16* __restrict__ V, const float* __restrict__ rmap,
    _Float16* __restrict__ O, int layer) {
  extern __shared__ __align__(16) char smem[];
  _Float16* Qs = (_Float16*)smem;
  _Float16* Ks = (_Float16*)(smem + SM_QS);
  _Float16* Vt = (_Float16*)(smem + SM_QS + SM_KS);
  float*    S  = (float*)(smem + SM_QS + SM_KS + SM_VT);

  const int h = blockIdx.x, b = blockIdx.y;
  const int tid = threadIdx.x;
  const int lane = tid & 31;
  const int wid = tid >> 5;
  const int hlf = lane >> 4;
  const int l16 = lane & 15;
  const int kb = hlf * 8;
  const size_t base = (size_t)(b * TPAD) * DMODEL + h * DHEAD;

  // Zero pad rows of Qs/Ks and pad cols of Vt (disjoint from async targets).
  for (int idx = tid; idx < (TPAD - TSEQ) * DHEAD; idx += 256) {
    int t = TSEQ + (idx >> 6), c = idx & 63;
    Qs[t * QS_STRIDE + c] = (_Float16)0;
    Ks[t * QS_STRIDE + c] = (_Float16)0;
    Vt[c * VT_STRIDE + t] = (_Float16)0;
  }
  // Async-copy valid Q/K rows (contiguous 16B segments).
  for (int idx = tid; idx < TSEQ * 8; idx += 256) {
    int t = idx >> 3, s8 = (idx & 7) * 8;
    async_cp16(lds_off(Qs + t * QS_STRIDE + s8), Q + base + (size_t)t * DMODEL + s8);
    async_cp16(lds_off(Ks + t * QS_STRIDE + s8), Kmat + base + (size_t)t * DMODEL + s8);
  }
  // V transposed (scattered) - manual.
  for (int idx = tid; idx < TSEQ * DHEAD; idx += 256) {
    int t = idx >> 6, c = idx & 63;
    Vt[c * VT_STRIDE + t] = V[base + (size_t)t * DMODEL + c];
  }
  wait_async0();
  __syncthreads();

  // Phase 1: S = (Q @ K^T) * rsqrt(DK). Bt of QK^T == K rows as stored.
  for (int job = wid; job < (TPAD / 16) * (TPAD / 16); job += 8) {
    int tm = job / (TPAD / 16), tn = job % (TPAD / 16);
    v8f acc = {};
    const _Float16* qrow = Qs + (tm * 16 + l16) * QS_STRIDE;
    const _Float16* krow = Ks + (tn * 16 + l16) * QS_STRIDE;
#pragma unroll
    for (int kk = 0; kk < DHEAD / 32; ++kk)
      acc = wmma_f16(frag_ld_h(qrow, kk * 32 + kb),
                     frag_ld_h(krow, kk * 32 + kb), acc);
#pragma unroll
    for (int v = 0; v < 8; ++v)
      S[(tm * 16 + hlf * 8 + v) * S_STRIDE + tn * 16 + l16] = acc[v] * 0.125f;
  }
  __syncthreads();

  // Phase 2: per row: softmax -> 5x logistic map -> softmax. One wave per row.
  const float rv = rmap[layer * NHEAD + h];
  for (int row = wid; row < TSEQ; row += 8) {
    float* Srow = S + row * S_STRIDE;
    float vals[7];
    float mx = -1e30f;
#pragma unroll
    for (int i = 0; i < 7; ++i) {
      int n = lane + i * 32;
      float xv = (n < TSEQ) ? Srow[n] : -1e30f;
      vals[i] = xv;
      mx = fmaxf(mx, xv);
    }
#pragma unroll
    for (int d = 16; d >= 1; d >>= 1) mx = fmaxf(mx, __shfl_xor(mx, d));
    float sum = 0.f;
#pragma unroll
    for (int i = 0; i < 7; ++i) { float e = expf(vals[i] - mx); vals[i] = e; sum += e; }
#pragma unroll
    for (int d = 16; d >= 1; d >>= 1) sum += __shfl_xor(sum, d);
    float inv = 1.0f / sum;
#pragma unroll
    for (int i = 0; i < 7; ++i) {
      float p = vals[i] * inv;
#pragma unroll
      for (int it = 0; it < CHAOS_ITERS; ++it) p = rv * p * (1.0f - p);
      int n = lane + i * 32;
      vals[i] = (n < TSEQ) ? p : -1e30f;
    }
    float mx2 = -1e30f;
#pragma unroll
    for (int i = 0; i < 7; ++i) mx2 = fmaxf(mx2, vals[i]);
#pragma unroll
    for (int d = 16; d >= 1; d >>= 1) mx2 = fmaxf(mx2, __shfl_xor(mx2, d));
    float sum2 = 0.f;
#pragma unroll
    for (int i = 0; i < 7; ++i) { float e = expf(vals[i] - mx2); vals[i] = e; sum2 += e; }
#pragma unroll
    for (int d = 16; d >= 1; d >>= 1) sum2 += __shfl_xor(sum2, d);
    float inv2 = 1.0f / sum2;
#pragma unroll
    for (int i = 0; i < 7; ++i) {
      int n = lane + i * 32;
      if (n < TPAD) Srow[n] = (n < TSEQ) ? vals[i] * inv2 : 0.f;
    }
  }
  __syncthreads();

  // Phase 3: O = P @ V (P fragments assembled f32->f16 from LDS).
  for (int job = wid; job < (TPAD / 16) * (DHEAD / 16); job += 8) {
    int tm = job >> 2, tn = job & 3;
    v8f acc = {};
    const float* prow = S + (tm * 16 + l16) * S_STRIDE;
    const _Float16* vrow = Vt + (tn * 16 + l16) * VT_STRIDE;
#pragma unroll
    for (int kk = 0; kk < TPAD / 32; ++kk)
      acc = wmma_f16(frag_ld_f32(prow, kk * 32 + kb),
                     frag_ld_h(vrow, kk * 32 + kb), acc);
#pragma unroll
    for (int v = 0; v < 8; ++v)
      O[base + (size_t)(tm * 16 + hlf * 8 + v) * DMODEL + tn * 16 + l16] =
          (_Float16)acc[v];
  }
}

// --- LayerNorm: one block per row; writes f16 (zeros for padded rows) -------
__global__ __launch_bounds__(256) void ln_rows(
    const float* __restrict__ x, const float* __restrict__ gs,
    const float* __restrict__ gb, _Float16* __restrict__ y) {
  __shared__ float red[256];
  const int row = blockIdx.x;
  const int t = row % TPAD;
  const int tid = threadIdx.x;
  _Float16* yr = y + (size_t)row * DMODEL;
  if (t >= TSEQ) {
    for (int c = tid; c < DMODEL; c += 256) yr[c] = (_Float16)0;
    return;
  }
  const float* xr = x + (size_t)row * DMODEL;
  float sm = 0.f;
  for (int c = tid; c < DMODEL; c += 256) sm += xr[c];
  red[tid] = sm; __syncthreads();
  for (int s = 128; s > 0; s >>= 1) { if (tid < s) red[tid] += red[tid + s]; __syncthreads(); }
  float mu = red[0] / DMODEL; __syncthreads();
  float vs = 0.f;
  for (int c = tid; c < DMODEL; c += 256) { float d = xr[c] - mu; vs += d * d; }
  red[tid] = vs; __syncthreads();
  for (int s = 128; s > 0; s >>= 1) { if (tid < s) red[tid] += red[tid + s]; __syncthreads(); }
  float rstd = rsqrtf(red[0] / DMODEL + LN_EPS);
  for (int c = tid; c < DMODEL; c += 256)
    yr[c] = (_Float16)((xr[c] - mu) * rstd * gs[c] + gb[c]);
}

// --- Head: LN over patch tokens then dot with head_w ------------------------
__global__ __launch_bounds__(256) void head_kernel(
    const float* __restrict__ hbuf, const float* __restrict__ gs,
    const float* __restrict__ gb, const float* __restrict__ hw,
    const float* __restrict__ hb, float* __restrict__ out) {
  __shared__ float red[256];
  const int row = blockIdx.x;  // 0..B*NPATCH-1
  const int b = row / NPATCH, n = row % NPATCH;
  const int tid = threadIdx.x;
  const float* xr = hbuf + (size_t)(b * TPAD + 1 + n) * DMODEL;
  float sm = 0.f;
  for (int c = tid; c < DMODEL; c += 256) sm += xr[c];
  red[tid] = sm; __syncthreads();
  for (int s = 128; s > 0; s >>= 1) { if (tid < s) red[tid] += red[tid + s]; __syncthreads(); }
  float mu = red[0] / DMODEL; __syncthreads();
  float vs = 0.f;
  for (int c = tid; c < DMODEL; c += 256) { float d = xr[c] - mu; vs += d * d; }
  red[tid] = vs; __syncthreads();
  for (int s = 128; s > 0; s >>= 1) { if (tid < s) red[tid] += red[tid + s]; __syncthreads(); }
  float rstd = rsqrtf(red[0] / DMODEL + LN_EPS); __syncthreads();
  float dot = 0.f;
  for (int c = tid; c < DMODEL; c += 256)
    dot += ((xr[c] - mu) * rstd * gs[c] + gb[c]) * hw[c];
  red[tid] = dot; __syncthreads();
  for (int s = 128; s > 0; s >>= 1) { if (tid < s) red[tid] += red[tid + s]; __syncthreads(); }
  if (tid == 0) out[row] = red[0] + hb[0];
}

// --- im2col (f32 -> f16), zero-padded to MPATCH_PAD rows --------------------
__global__ void im2col_patches(const float* __restrict__ x, _Float16* __restrict__ Xc) {
  size_t i = (size_t)blockIdx.x * blockDim.x + threadIdx.x;
  const size_t total = (size_t)MPATCH_PAD * KPATCH;
  if (i >= total) return;
  int col = (int)(i % KPATCH);
  int rowp = (int)(i / KPATCH);
  if (rowp >= BATCH * NPATCH) { Xc[i] = (_Float16)0; return; }
  int b = rowp / NPATCH, pidx = rowp % NPATCH;
  int gy = pidx / 14, gx = pidx % 14;
  int c = col >> 8, py = (col >> 4) & 15, px = col & 15;
  Xc[i] = (_Float16)x[(((size_t)b * 3 + c) * 224 + (gy * 16 + py)) * 224 + gx * 16 + px];
}

// --- assemble h0 = [cls | tok] + pos, zero padded rows ----------------------
__global__ void assemble_h0(const float* __restrict__ tok, const float* __restrict__ cls,
                            const float* __restrict__ pos, float* __restrict__ hbuf) {
  size_t i = (size_t)blockIdx.x * blockDim.x + threadIdx.x;
  if (i >= (size_t)MTOK * DMODEL) return;
  int d = (int)(i % DMODEL);
  int row = (int)(i / DMODEL);
  int b = row / TPAD, t = row % TPAD;
  float v;
  if (t == 0) v = cls[d] + pos[d];
  else if (t < TSEQ) v = tok[(size_t)(b * NPATCH + t - 1) * DMODEL + d] + pos[(size_t)t * DMODEL + d];
  else v = 0.f;
  hbuf[i] = v;
}

// --- f32 [K][N] -> f16 transposed [N][K] (weights) --------------------------
__global__ void cvt_transpose_f16(const float* __restrict__ src, _Float16* __restrict__ dst,
                                  int K, int N) {
  size_t i = (size_t)blockIdx.x * blockDim.x + threadIdx.x;
  if (i >= (size_t)K * N) return;
  int n = (int)(i % N), k = (int)(i / N);
  dst[(size_t)n * K + k] = (_Float16)src[i];
}

// ---------------------------------------------------------------------------
extern "C" void kernel_launch(void* const* d_in, const int* in_sizes, int n_in,
                              void* d_out, int out_size, void* d_ws, size_t ws_size,
                              hipStream_t stream) {
  (void)in_sizes; (void)n_in; (void)out_size; (void)ws_size;
  const float* x       = (const float*)d_in[0];
  const float* patch_w = (const float*)d_in[1];
  const float* patch_b = (const float*)d_in[2];
  const float* cls_tok = (const float*)d_in[3];
  const float* pos_emb = (const float*)d_in[4];
  const float* ln1_s   = (const float*)d_in[5];
  const float* ln1_b   = (const float*)d_in[6];
  const float* wq = (const float*)d_in[7];  const float* bq = (const float*)d_in[8];
  const float* wk = (const float*)d_in[9];  const float* bk = (const float*)d_in[10];
  const float* wv = (const float*)d_in[11]; const float* bv = (const float*)d_in[12];
  const float* wo = (const float*)d_in[13]; const float* bo = (const float*)d_in[14];
  const float* rmap = (const float*)d_in[15];
  const float* ln2_s = (const float*)d_in[16];
  const float* ln2_b = (const float*)d_in[17];
  const float* w1 = (const float*)d_in[18]; const float* b1 = (const float*)d_in[19];
  const float* w2 = (const float*)d_in[20]; const float* b2 = (const float*)d_in[21];
  const float* lnh_s = (const float*)d_in[22];
  const float* lnh_b = (const float*)d_in[23];
  const float* head_w = (const float*)d_in[24];
  const float* head_b = (const float*)d_in[25];
  float* out = (float*)d_out;

  // Workspace carve-out (~90 MB total).
  char* wp = (char*)d_ws;
  auto take = [&](size_t bytes) -> char* {
    char* p = wp; wp += (bytes + 255) & ~(size_t)255; return p;
  };
  _Float16* Xc  = (_Float16*)take((size_t)MPATCH_PAD * KPATCH * 2);
  float*    tok = (float*)take((size_t)MPATCH_PAD * DMODEL * 4);
  float*    hb  = (float*)take((size_t)MTOK * DMODEL * 4);
  _Float16* y   = (_Float16*)take((size_t)MTOK * DMODEL * 2);
  _Float16* qb  = (_Float16*)take((size_t)MTOK * DMODEL * 2);
  _Float16* kbf = (_Float16*)take((size_t)MTOK * DMODEL * 2);
  _Float16* vb  = (_Float16*)take((size_t)MTOK * DMODEL * 2);
  _Float16* ao  = (_Float16*)take((size_t)MTOK * DMODEL * 2);
  _Float16* u   = (_Float16*)take((size_t)MTOK * DMLP * 2);
  _Float16* wf  = (_Float16*)take((size_t)DMODEL * DMLP * 2);  // f16 Bt scratch

  auto cvtT = [&](const float* s, _Float16* d, int K, int N) {
    size_t tot = (size_t)K * N;
    cvt_transpose_f16<<<(unsigned)((tot + 255) / 256), 256, 0, stream>>>(s, d, K, N);
  };
  auto gemm = [&](const _Float16* A, int lda, const _Float16* Bt, int ldb,
                  const float* bias, const float* resid, float* c32, _Float16* c16,
                  int M, int N, int K, int flags) {
    gemm_wmma<<<dim3(N / GEMM_TN, M / GEMM_TM), 256, 0, stream>>>(
        A, lda, Bt, ldb, bias, resid, c32, c16, M, N, K, flags);
  };

  // Patch embedding
  {
    size_t tot = (size_t)MPATCH_PAD * KPATCH;
    im2col_patches<<<(unsigned)((tot + 255) / 256), 256, 0, stream>>>(x, Xc);
  }
  cvtT(patch_w, wf, KPATCH, DMODEL);
  gemm(Xc, KPATCH, wf, KPATCH, patch_b, nullptr, tok, nullptr,
       MPATCH_PAD, DMODEL, KPATCH, 0);
  {
    size_t tot = (size_t)MTOK * DMODEL;
    assemble_h0<<<(unsigned)((tot + 255) / 256), 256, 0, stream>>>(tok, cls_tok, pos_emb, hb);
  }

  for (int i = 0; i < NLAYER; ++i) {
    const size_t wDD = (size_t)DMODEL * DMODEL;
    const size_t wDM = (size_t)DMODEL * DMLP;
    ln_rows<<<MTOK, 256, 0, stream>>>(hb, ln1_s + i * DMODEL, ln1_b + i * DMODEL, y);
    cvtT(wq + (size_t)i * wDD, wf, DMODEL, DMODEL);
    gemm(y, DMODEL, wf, DMODEL, bq + i * DMODEL, nullptr, nullptr, qb, MTOK, DMODEL, DMODEL, 0);
    cvtT(wk + (size_t)i * wDD, wf, DMODEL, DMODEL);
    gemm(y, DMODEL, wf, DMODEL, bk + i * DMODEL, nullptr, nullptr, kbf, MTOK, DMODEL, DMODEL, 0);
    cvtT(wv + (size_t)i * wDD, wf, DMODEL, DMODEL);
    gemm(y, DMODEL, wf, DMODEL, bv + i * DMODEL, nullptr, nullptr, vb, MTOK, DMODEL, DMODEL, 0);

    attn_chaos<<<dim3(NHEAD, BATCH), 256, SMEM_ATTN, stream>>>(qb, kbf, vb, rmap, ao, i);

    cvtT(wo + (size_t)i * wDD, wf, DMODEL, DMODEL);
    gemm(ao, DMODEL, wf, DMODEL, bo + i * DMODEL, hb, hb, nullptr, MTOK, DMODEL, DMODEL, 0);

    ln_rows<<<MTOK, 256, 0, stream>>>(hb, ln2_s + i * DMODEL, ln2_b + i * DMODEL, y);
    cvtT(w1 + (size_t)i * wDM, wf, DMODEL, DMLP);
    gemm(y, DMODEL, wf, DMODEL, b1 + i * DMLP, nullptr, nullptr, u, MTOK, DMLP, DMODEL, FLAG_GELU);
    cvtT(w2 + (size_t)i * wDM, wf, DMLP, DMODEL);
    gemm(u, DMLP, wf, DMLP, b2 + i * DMODEL, hb, hb, nullptr, MTOK, DMODEL, DMLP, 0);
  }

  head_kernel<<<BATCH * NPATCH, 256, 0, stream>>>(hb, lnh_s, lnh_b, head_w, head_b, out);
}
